// SrgcnHead_11879879541098
// MI455X (gfx1250) — compile-verified, matrix operand order checked by reading, not run on hardware
//
#include <hip/hip_runtime.h>
#include <hip/hip_bf16.h>
#include <math.h>

#define ALPHA 0.2f

typedef __attribute__((ext_vector_type(16))) _Float16 v16h;
typedef __attribute__((ext_vector_type(8)))  _Float16 v8h;
typedef __attribute__((ext_vector_type(8)))  float    v8f;

// ---------- helpers ----------
__device__ __forceinline__ unsigned fenc(float f) {
    unsigned u = __float_as_uint(f);
    return (u & 0x80000000u) ? ~u : (u | 0x80000000u);
}
__device__ __forceinline__ float fdec(unsigned k) {
    unsigned u = (k & 0x80000000u) ? (k & 0x7FFFFFFFu) : ~k;
    return __uint_as_float(u);
}
__device__ __forceinline__ float leaky(float s) { return s > 0.f ? s : ALPHA * s; }
__device__ __forceinline__ void atomAddF(float* p, float v) {
    __hip_atomic_fetch_add(p, v, __ATOMIC_RELAXED, __HIP_MEMORY_SCOPE_AGENT);
}

// ---------- conversion kernels ----------
__global__ void cvt_x_f16(const float* __restrict__ x, _Float16* __restrict__ xb, int n) {
    int i = blockIdx.x * blockDim.x + threadIdx.x;
    if (i < n) xb[i] = (_Float16)x[i];
}

// W: [2,128,64] f32 row-major -> WbT: [2,64,128] f16 (transposed per subhead)
__global__ void cvt_w_f16(const float* __restrict__ W, _Float16* __restrict__ WbT) {
    int i = blockIdx.x * blockDim.x + threadIdx.x;
    if (i >= 2 * 64 * 128) return;
    int sub = i >> 13;          // / 8192
    int r   = i & 8191;
    int d   = r >> 7;           // / 128
    int k   = r & 127;
    WbT[i] = (_Float16)W[sub * 8192 + k * 64 + d];
}

// ---------- WMMA GEMM: h[N,64] = xb[N,128] @ W[128,64] ----------
// one wave per 16x16 output tile, K=128 in 4 steps of 32
__global__ void gemm_wmma(const _Float16* __restrict__ xb,
                          const _Float16* __restrict__ WbT,   // [64,128] (col-of-W contiguous)
                          float* __restrict__ h, int nRowTiles) {
    int wave = (blockIdx.x * blockDim.x + threadIdx.x) >> 5;
    int lane = threadIdx.x & 31;
    int nTiles = nRowTiles * 4;                 // 64/16 = 4 col tiles
    if (wave >= nTiles) return;
    int rowBase = (wave >> 2) * 16;
    int colBase = (wave & 3) * 16;
    int mn   = lane & 15;                       // M for A, N for B
    int half = lane >> 4;                       // K-half select
    const _Float16* aRow = xb  + (size_t)(rowBase + mn) * 128 + half * 8;
    const _Float16* bRow = WbT + (size_t)(colBase + mn) * 128 + half * 8;
    v8f c = {};
#pragma unroll
    for (int kk = 0; kk < 128; kk += 32) {
        v8h a_lo = *(const v8h*)(aRow + kk);
        v8h a_hi = *(const v8h*)(aRow + kk + 16);
        v8h b_lo = *(const v8h*)(bRow + kk);
        v8h b_hi = *(const v8h*)(bRow + kk + 16);
        v16h A = __builtin_shufflevector(a_lo, a_hi, 0,1,2,3,4,5,6,7,8,9,10,11,12,13,14,15);
        v16h B = __builtin_shufflevector(b_lo, b_hi, 0,1,2,3,4,5,6,7,8,9,10,11,12,13,14,15);
        c = __builtin_amdgcn_wmma_f32_16x16x32_f16(false, A, false, B, (short)0, c,
                                                   false, false);
    }
    // D layout: lane -> N = lane&15 ; VGPR r -> M = r + 8*(lane>>4)
    float* out = h + (size_t)(rowBase + half * 8) * 64 + colBase + mn;
#pragma unroll
    for (int r = 0; r < 8; ++r) out[(size_t)r * 64] = c[r];
}

// ---------- per-node attention scores (wave per node) ----------
__global__ void scores_k(const float* __restrict__ h, const float* __restrict__ a_src,
                         const float* __restrict__ a_dst, float* __restrict__ s_r,
                         float* __restrict__ s_c, int n) {
    int node = (blockIdx.x * blockDim.x + threadIdx.x) >> 5;
    int lane = threadIdx.x & 31;
    if (node >= n) return;
    float2 hv = *(const float2*)(h + (size_t)node * 64 + lane * 2);
    float2 as = *(const float2*)(a_src + lane * 2);
    float2 ad = *(const float2*)(a_dst + lane * 2);
    float pr = hv.x * as.x + hv.y * as.y;
    float pc = hv.x * ad.x + hv.y * ad.y;
#pragma unroll
    for (int off = 16; off; off >>= 1) {
        pr += __shfl_xor(pr, off, 32);
        pc += __shfl_xor(pc, off, 32);
    }
    if (lane == 0) { s_r[node] = pr; s_c[node] = pc; }
}

// ---------- edge softmax pass 1: segment max (ordered-uint atomicMax) ----------
__global__ void edge_max_k(const int* __restrict__ ei, const float* __restrict__ s_r,
                           const float* __restrict__ s_c, unsigned* __restrict__ menc, int E) {
    int e = blockIdx.x * blockDim.x + threadIdx.x;
    if (e >= E) return;
    int row = ei[e], col = ei[E + e];
    float ev = leaky(s_r[row] + s_c[col]);
    atomicMax(menc + row, fenc(ev));
}

__global__ void decode_k(unsigned* __restrict__ m, int n) {
    int i = blockIdx.x * blockDim.x + threadIdx.x;
    if (i < n) { float f = fdec(m[i]); ((float*)m)[i] = f; }
}

// ---------- edge softmax pass 2: segment sum of exp ----------
__global__ void edge_sum_k(const int* __restrict__ ei, const float* __restrict__ s_r,
                           const float* __restrict__ s_c, const float* __restrict__ m,
                           float* __restrict__ denom, int E) {
    int e = blockIdx.x * blockDim.x + threadIdx.x;
    if (e >= E) return;
    int row = ei[e], col = ei[E + e];
    float ev = leaky(s_r[row] + s_c[col]);
    atomAddF(denom + row, expf(ev - m[row]));
}

// ---------- SpMM hop: vout[row] += att(e) * vin[col]  (wave per edge) ----------
__global__ void spmm_k(const int* __restrict__ ei, const float* __restrict__ s_r,
                       const float* __restrict__ s_c, const float* __restrict__ m,
                       const float* __restrict__ denom, const float* __restrict__ vin,
                       float* __restrict__ vout, int E) {
    int e = (blockIdx.x * blockDim.x + threadIdx.x) >> 5;
    int lane = threadIdx.x & 31;
    if (e >= E) return;
    int row = ei[e], col = ei[E + e];
    float ev = leaky(s_r[row] + s_c[col]);
    float val = expf(ev - m[row]) / denom[row];
    float2 hv = *(const float2*)(vin + (size_t)col * 64 + lane * 2);
    float* dst = vout + (size_t)row * 64 + lane * 2;
    atomAddF(dst,     val * hv.x);
    atomAddF(dst + 1, val * hv.y);
}

// ---------- finalize: bias + NodeAdaptiveEncoder + leaky (wave per node) ----------
__global__ void final_k(const float* __restrict__ vh, const float* __restrict__ b,
                        const float* __restrict__ fc, const float* __restrict__ bf,
                        float* __restrict__ out, int n, int sub) {
    int node = (blockIdx.x * blockDim.x + threadIdx.x) >> 5;
    int lane = threadIdx.x & 31;
    if (node >= n) return;
    float2 v = *(const float2*)(vh + (size_t)node * 64 + lane * 2);
    if (v.x != v.x) v.x = 0.f;
    if (v.y != v.y) v.y = 0.f;
    float2 bb = *(const float2*)(b + lane * 2);
    v.x += bb.x; v.y += bb.y;
    float2 f = *(const float2*)(fc + lane * 2);
    float p = v.x * f.x + v.y * f.y;
#pragma unroll
    for (int off = 16; off; off >>= 1) p += __shfl_xor(p, off, 32);
    float g = 1.f / (1.f + expf(-(p + bf[0])));
    float r0 = fmaxf(v.x, 0.f) + g * fminf(v.x, 0.f);
    float r1 = fmaxf(v.y, 0.f) + g * fminf(v.y, 0.f);
    float2 res; res.x = leaky(r0); res.y = leaky(r1);
    *(float2*)(out + (size_t)node * 128 + sub * 64 + lane * 2) = res;
}

extern "C" void kernel_launch(void* const* d_in, const int* in_sizes, int n_in,
                              void* d_out, int out_size, void* d_ws, size_t ws_size,
                              hipStream_t stream) {
    const float* x     = (const float*)d_in[0];
    const int*   ei    = (const int*)d_in[1];
    const float* W     = (const float*)d_in[2];
    const float* b     = (const float*)d_in[3];
    const float* fc    = (const float*)d_in[4];
    const float* bf    = (const float*)d_in[5];
    const float* a_src = (const float*)d_in[6];
    const float* a_dst = (const float*)d_in[7];
    float* out = (float*)d_out;

    int N = in_sizes[0] / 128;
    int E = in_sizes[1] / 2;

    char* ws = (char*)d_ws;
    size_t off = 0;
    auto alloc = [&](size_t bytes) -> void* {
        void* p = ws + off;
        off = (off + bytes + 255) & ~(size_t)255;
        return p;
    };
    _Float16* xb   = (_Float16*)alloc((size_t)N * 128 * 2);
    _Float16* WbT  = (_Float16*)alloc((size_t)2 * 64 * 128 * 2);
    float*    h    = (float*)alloc((size_t)N * 64 * 4);
    float*    bufA = (float*)alloc((size_t)N * 64 * 4);
    float*    bufB = (float*)alloc((size_t)N * 64 * 4);
    float*    s_r  = (float*)alloc((size_t)N * 4);
    float*    s_c  = (float*)alloc((size_t)N * 4);
    unsigned* menc = (unsigned*)alloc((size_t)N * 4);
    float*    den  = (float*)alloc((size_t)N * 4);

    cvt_x_f16<<<(N * 128 + 255) / 256, 256, 0, stream>>>(x, xb, N * 128);
    cvt_w_f16<<<(2 * 64 * 128 + 255) / 256, 256, 0, stream>>>(W, WbT);

    int nRowTiles = N / 16;
    int nTiles = nRowTiles * 4;
    dim3 blk(256);
    int gemmBlocks  = (nTiles + 7) / 8;   // 8 waves per block
    int nodeWBlocks = (N + 7) / 8;        // wave-per-node kernels
    int edgeTBlocks = (E + 255) / 256;    // thread-per-edge kernels
    int edgeWBlocks = (E + 7) / 8;        // wave-per-edge kernels

    for (int sub = 0; sub < 2; ++sub) {
        gemm_wmma<<<gemmBlocks, blk, 0, stream>>>(xb, WbT + (size_t)sub * 64 * 128, h, nRowTiles);
        scores_k<<<nodeWBlocks, blk, 0, stream>>>(h, a_src + sub * 64, a_dst + sub * 64,
                                                  s_r, s_c, N);
        hipMemsetAsync(menc, 0, (size_t)N * 4, stream);
        edge_max_k<<<edgeTBlocks, blk, 0, stream>>>(ei, s_r, s_c, menc, E);
        decode_k<<<(N + 255) / 256, blk, 0, stream>>>(menc, N);
        hipMemsetAsync(den, 0, (size_t)N * 4, stream);
        edge_sum_k<<<edgeTBlocks, blk, 0, stream>>>(ei, s_r, s_c, (const float*)menc, den, E);

        hipMemsetAsync(bufA, 0, (size_t)N * 64 * 4, stream);
        spmm_k<<<edgeWBlocks, blk, 0, stream>>>(ei, s_r, s_c, (const float*)menc, den,
                                                h, bufA, E);
        const float* vfin = bufA;
        if (sub == 1) {
            hipMemsetAsync(bufB, 0, (size_t)N * 64 * 4, stream);
            spmm_k<<<edgeWBlocks, blk, 0, stream>>>(ei, s_r, s_c, (const float*)menc, den,
                                                    bufA, bufB, E);
            vfin = bufB;
        }
        final_k<<<nodeWBlocks, blk, 0, stream>>>(vfin, b + sub * 64, fc + sub * 64,
                                                 bf + sub, out, N, sub);
    }
}
